// ComplexMultiheadAttention_54425825575493
// MI455X (gfx1250) — matrix operand order repaired and checked by reading, not compiled, not run on hardware
//
#include <hip/hip_runtime.h>
#include <hip/hip_bf16.h>

// ---------------------------------------------------------------------------
// Complex multi-head attention for MI455X (gfx1250), wave32 + WMMA bf16
// + async global->LDS staging (ASYNCcnt) for shared K tiles.
// ---------------------------------------------------------------------------

typedef __bf16 bf16;
typedef __attribute__((ext_vector_type(16))) __bf16 v16bf;
typedef __attribute__((ext_vector_type(8)))  float  v8f;
typedef __attribute__((ext_vector_type(4)))  unsigned int   u32x4;
typedef __attribute__((ext_vector_type(4)))  unsigned short u16x4;

constexpr int Bb  = 2;
constexpr int Ll  = 2048;
constexpr int Dd  = 1024;
constexpr int NHh = 16;
constexpr int HDd = 64;
constexpr int Mm  = Bb * Ll;                  // 4096 rows
constexpr size_t MD = (size_t)Mm * Dd;        // 4M elems
constexpr size_t DD = (size_t)Dd * Dd;        // 1M elems

union FragU { v16bf v; u32x4 q[2]; unsigned int u[8]; };

__device__ inline float fast_exp2(float x) { return __builtin_amdgcn_exp2f(x); }

// A-matrix 16x32 bf16 fragment (ISA 7.12.2): lane l<16 -> row l, K {0..7,16..23};
// lane l+16 -> row l, K {8..15,24..31}. Two 16B chunks per lane.
__device__ inline v16bf load_frag_a(const bf16* __restrict__ tile, int ld, int lane) {
    int row = lane & 15;
    int k   = (lane & 16) ? 8 : 0;
    const bf16* p = tile + (size_t)row * ld + k;
    FragU f;
    f.q[0] = *(const u32x4*)(p);
    f.q[1] = *(const u32x4*)(p + 16);
    return f.v;
}

// B-matrix 32x16 bf16 fragment from "Bt" storage where Bt[n][k] = B[k][n]
// (a natural row-major matrix whose rows are the N columns; B is loaded
// column-major per lane). lane l<16 -> col n=l, K {0..7,16..23};
// lane l+16 -> col n=l, K {8..15,24..31}. Two 16B chunks per lane.
__device__ inline v16bf load_frag_b(const bf16* __restrict__ bt, int ld, int lane) {
    int n = lane & 15;
    int k = (lane & 16) ? 8 : 0;
    const bf16* p = bt + (size_t)n * ld + k;
    FragU f;
    f.q[0] = *(const u32x4*)(p);
    f.q[1] = *(const u32x4*)(p + 16);
    return f.v;
}

// bf16 WMMA forbids A/B NEG (ISA: NEG = {CNeg,0,0}) -> flip sign bits in VALU.
__device__ inline v16bf neg16(v16bf x) {
    FragU f; f.v = x;
#pragma unroll
    for (int i = 0; i < 8; ++i) f.u[i] ^= 0x80008000u;
    return f.v;
}

__device__ inline v8f wmma_bf(v16bf a, v16bf b, v8f c) {
    return __builtin_amdgcn_wmma_f32_16x16x32_bf16(
        false, a, false, b, (short)0, c, false, false);
}

// Async 16B global->LDS copy per lane (GLOBAL_LOAD_ASYNC_TO_LDS_B128,
// tracked by ASYNCcnt). The LDS destination VGPR carries the per-lane LDS
// byte address = low 32 bits of the generic pointer (aperture truncation).
__device__ inline void async_copy_b128(const bf16* gsrc, bf16* lds_dst) {
    const unsigned loff = (unsigned)(uintptr_t)lds_dst;
    const unsigned long long g = (unsigned long long)(uintptr_t)gsrc;
    asm volatile("global_load_async_to_lds_b128 %0, %1, off"
                 :: "v"(loff), "v"(g) : "memory");
}
__device__ inline void wait_asynccnt0() {
    asm volatile("s_wait_asynccnt 0x0" ::: "memory");
}

// ---------------------------------------------------------------------------
// fp32 -> bf16 conversion
// ---------------------------------------------------------------------------
__global__ __launch_bounds__(256) void cvt_f32_to_bf16(const float* __restrict__ s,
                                                       bf16* __restrict__ d, int n) {
    int i = (blockIdx.x * 256 + threadIdx.x) * 4;
    if (i + 3 < n) {
        float4 f = *(const float4*)(s + i);
        d[i + 0] = (bf16)f.x;
        d[i + 1] = (bf16)f.y;
        d[i + 2] = (bf16)f.z;
        d[i + 3] = (bf16)f.w;
    } else {
        for (; i < n; ++i) d[i] = (bf16)s[i];
    }
}

// ---------------------------------------------------------------------------
// Fused complex-linear GEMM (software-pipelined; see round-2 notes).
// ---------------------------------------------------------------------------
template <bool WRITE_F32>
__global__ __launch_bounds__(256) void clin_gemm(
    const bf16* __restrict__ Ar, const bf16* __restrict__ Ai,
    const bf16* __restrict__ Wr, const bf16* __restrict__ Wi,
    const float* __restrict__ br, const float* __restrict__ bi,
    void* __restrict__ outR_, void* __restrict__ outI_) {
    const int wave = threadIdx.x >> 5;
    const int lane = threadIdx.x & 31;
    const int row0 = blockIdx.x * 128 + wave * 16;
    const int col0 = blockIdx.y * 64;

    v8f accR[4] = {v8f{}, v8f{}, v8f{}, v8f{}};
    v8f accI[4] = {v8f{}, v8f{}, v8f{}, v8f{}};

    const bf16* ar = Ar + (size_t)row0 * Dd;
    const bf16* ai = Ai + (size_t)row0 * Dd;

    v16bf a_r = load_frag_a(ar, Dd, lane);
    v16bf a_i = load_frag_a(ai, Dd, lane);
    v16bf b_r[4], b_i[4];
#pragma unroll
    for (int n = 0; n < 4; ++n) {
        b_r[n] = load_frag_b(Wr + (size_t)(col0 + n * 16) * Dd, Dd, lane);
        b_i[n] = load_frag_b(Wi + (size_t)(col0 + n * 16) * Dd, Dd, lane);
    }

    for (int k0 = 0; k0 < Dd; k0 += 32) {
        const int kn = (k0 + 32 < Dd) ? (k0 + 32) : 0;   // wrap: harmless reload
        v16bf a_r_n = load_frag_a(ar + kn, Dd, lane);
        v16bf a_i_n = load_frag_a(ai + kn, Dd, lane);
        v16bf b_r_n[4], b_i_n[4];
#pragma unroll
        for (int n = 0; n < 4; ++n) {
            b_r_n[n] = load_frag_b(Wr + (size_t)(col0 + n * 16) * Dd + kn, Dd, lane);
            b_i_n[n] = load_frag_b(Wi + (size_t)(col0 + n * 16) * Dd + kn, Dd, lane);
        }
        __builtin_prefetch(ar + (size_t)(lane & 15) * Dd + kn, 0, 0);

        v16bf na_i = neg16(a_i);
#pragma unroll
        for (int n = 0; n < 4; ++n) {
            accR[n] = wmma_bf(a_r,  b_r[n], accR[n]);
            accR[n] = wmma_bf(na_i, b_i[n], accR[n]);
            accI[n] = wmma_bf(a_i,  b_r[n], accI[n]);
            accI[n] = wmma_bf(a_r,  b_i[n], accI[n]);
        }
        a_r = a_r_n; a_i = a_i_n;
#pragma unroll
        for (int n = 0; n < 4; ++n) { b_r[n] = b_r_n[n]; b_i[n] = b_i_n[n]; }
    }

#pragma unroll
    for (int n = 0; n < 4; ++n) {
        const int col = col0 + n * 16 + (lane & 15);
        const float brv = br[col], biv = bi[col];
        const float bR = brv - biv, bI = brv + biv;
#pragma unroll
        for (int j = 0; j < 8; ++j) {
            const int row = row0 + j + ((lane & 16) ? 8 : 0);
            const size_t off = (size_t)row * Dd + col;
            const float vr = accR[n][j] + bR;
            const float vi = accI[n][j] + bI;
            if constexpr (WRITE_F32) {
                ((float*)outR_)[off] = vr;
                ((float*)outI_)[off] = vi;
            } else {
                ((bf16*)outR_)[off] = (bf16)vr;
                ((bf16*)outI_)[off] = (bf16)vi;
            }
        }
    }
}

// ---------------------------------------------------------------------------
// Flash attention over complex scores, 64-key steps.
// K tiles (shared by all 8 waves) are staged once per block into LDS via
// GLOBAL_LOAD_ASYNC_TO_LDS_B128 (ASYNCcnt), removing the 8x-redundant global
// fragment reads. V is transposed into LDS; P round-trips per-wave LDS.
// ---------------------------------------------------------------------------
__global__ __launch_bounds__(256) void attn_kernel(
    const bf16* __restrict__ Qr, const bf16* __restrict__ Qi,
    const bf16* __restrict__ Kr, const bf16* __restrict__ Ki,
    const bf16* __restrict__ Vr, const bf16* __restrict__ Vi,
    bf16* __restrict__ Or, bf16* __restrict__ Oi) {
    __shared__ __align__(16) bf16 sKr[64 * HDd];       // K_r tile [key][d]
    __shared__ __align__(16) bf16 sKi[64 * HDd];       // K_i tile [key][d]
    __shared__ __align__(16) bf16 sVtr[HDd * 64];      // Vr^T tile [d][key]
    __shared__ __align__(16) bf16 sVti[HDd * 64];      // Vi^T tile [d][key]
    __shared__ __align__(16) bf16 sP[8 * 16 * 64];     // per-wave P [16][64]

    const int wave = threadIdx.x >> 5;
    const int lane = threadIdx.x & 31;
    const int bh = blockIdx.y;
    const int b = bh / NHh, h = bh % NHh;
    const int q0 = blockIdx.x * 128 + wave * 16;
    const float c2 = 0.125f * 1.44269504f;   // 1/sqrt(64) * log2(e)

    const size_t head_off = (size_t)h * HDd;
    const bf16* qr_t = Qr + ((size_t)(b * Ll + q0)) * Dd + head_off;
    const bf16* qi_t = Qi + ((size_t)(b * Ll + q0)) * Dd + head_off;

    const v16bf aqr0 = load_frag_a(qr_t,      Dd, lane);
    const v16bf aqr1 = load_frag_a(qr_t + 32, Dd, lane);
    const v16bf aqi0 = load_frag_a(qi_t,      Dd, lane);
    const v16bf aqi1 = load_frag_a(qi_t + 32, Dd, lane);

    v8f o_r[4] = {v8f{}, v8f{}, v8f{}, v8f{}};
    v8f o_i[4] = {v8f{}, v8f{}, v8f{}, v8f{}};
    float m[8], l[8];
#pragma unroll
    for (int j = 0; j < 8; ++j) { m[j] = -1e30f; l[j] = 0.0f; }

    bf16* pw = &sP[wave * (16 * 64)];
    const bf16* kr_base = Kr + ((size_t)(b * Ll)) * Dd + head_off;
    const bf16* ki_base = Ki + ((size_t)(b * Ll)) * Dd + head_off;
    const bf16* vr_base = Vr + ((size_t)(b * Ll)) * Dd + head_off;
    const bf16* vi_base = Vi + ((size_t)(b * Ll)) * Dd + head_off;

    for (int kk = 0; kk < Ll; kk += 64) {
        __syncthreads();   // previous step's LDS consumers are done

        // ---- async-stage K tiles: 512 x 16B chunks per matrix ----
        for (int c = threadIdx.x; c < 512; c += 256) {
            const int key = c >> 3;            // 0..63
            const int dc  = c & 7;             // 16B chunk within the row
            const size_t goff = (size_t)(kk + key) * Dd + dc * 8;
            async_copy_b128(kr_base + goff, &sKr[c * 8]);
            async_copy_b128(ki_base + goff, &sKi[c * 8]);
        }

        // ---- stage V^T tile: b64 row loads -> packed b32 transposed stores ----
        for (int idx = threadIdx.x; idx < 1024; idx += 256) {
            const int mat = idx >> 9;          // 0: real, 1: imag
            const int t   = idx & 511;
            const int kp  = t & 31;            // key pair 0..31
            const int dc  = t >> 5;            // d chunk 0..15
            const int key0 = kp * 2;
            const int d0   = dc * 4;
            const bf16* src = mat ? vi_base : vr_base;
            bf16* dst       = mat ? sVti : sVtr;
            const u16x4 a = *(const u16x4*)(src + (size_t)(kk + key0)     * Dd + d0);
            const u16x4 c = *(const u16x4*)(src + (size_t)(kk + key0 + 1) * Dd + d0);
#pragma unroll
            for (int j = 0; j < 4; ++j) {
                const unsigned int w = (unsigned int)a[j] | ((unsigned int)c[j] << 16);
                *(unsigned int*)(dst + (d0 + j) * 64 + key0) = w;
            }
        }
        wait_asynccnt0();   // this wave's async copies landed in LDS
        __syncthreads();    // publish K + V^T tiles across the workgroup

        // ---- S = Qr Kr^T + Qi Ki^T : 16 rows x 64 keys (4 tiles), K from LDS ----
        v8f s[4];
#pragma unroll
        for (int n = 0; n < 4; ++n) {
            const bf16* krn = &sKr[(n * 16) * HDd];
            const bf16* kin = &sKi[(n * 16) * HDd];
            const v16bf bkr0 = load_frag_b(krn,      HDd, lane);
            const v16bf bkr1 = load_frag_b(krn + 32, HDd, lane);
            const v16bf bki0 = load_frag_b(kin,      HDd, lane);
            const v16bf bki1 = load_frag_b(kin + 32, HDd, lane);
            v8f acc = v8f{};
            acc = wmma_bf(aqr0, bkr0, acc);
            acc = wmma_bf(aqr1, bkr1, acc);
            acc = wmma_bf(aqi0, bki0, acc);
            acc = wmma_bf(aqi1, bki1, acc);
            s[n] = acc;
        }

        // ---- online softmax (exp2 domain); half-wave row reductions ----
        float alpha[8];
#pragma unroll
        for (int j = 0; j < 8; ++j) {
            float t = fmaxf(fmaxf(s[0][j], s[1][j]), fmaxf(s[2][j], s[3][j])) * c2;
#pragma unroll
            for (int mk = 1; mk < 16; mk <<= 1)
                t = fmaxf(t, __shfl_xor(t, mk, 32));
            const float mnew = fmaxf(m[j], t);
            alpha[j] = fast_exp2(m[j] - mnew);
            m[j] = mnew;
        }
#pragma unroll
        for (int j = 0; j < 8; ++j) {
            const float p0 = fast_exp2(s[0][j] * c2 - m[j]);
            const float p1 = fast_exp2(s[1][j] * c2 - m[j]);
            const float p2 = fast_exp2(s[2][j] * c2 - m[j]);
            const float p3 = fast_exp2(s[3][j] * c2 - m[j]);
            float rs = (p0 + p1) + (p2 + p3);
#pragma unroll
            for (int mk = 1; mk < 16; mk <<= 1)
                rs += __shfl_xor(rs, mk, 32);
            l[j] = l[j] * alpha[j] + rs;
            const int row = j + ((lane & 16) ? 8 : 0);
            const int col = lane & 15;
            pw[row * 64 + col]      = (bf16)p0;
            pw[row * 64 + col + 16] = (bf16)p1;
            pw[row * 64 + col + 32] = (bf16)p2;
            pw[row * 64 + col + 48] = (bf16)p3;
        }
#pragma unroll
        for (int n = 0; n < 4; ++n)
#pragma unroll
            for (int j = 0; j < 8; ++j) {
                o_r[n][j] *= alpha[j];
                o_i[n][j] *= alpha[j];
            }

        // ---- O += P x V ; P re-read as A-fragments from LDS ----
        const v16bf ap0 = load_frag_a(pw,      64, lane);   // keys 0..31
        const v16bf ap1 = load_frag_a(pw + 32, 64, lane);   // keys 32..63
#pragma unroll
        for (int n = 0; n < 4; ++n) {
            const v16bf bvr0 = load_frag_b(&sVtr[n * 16 * 64],      64, lane);
            const v16bf bvr1 = load_frag_b(&sVtr[n * 16 * 64 + 32], 64, lane);
            const v16bf bvi0 = load_frag_b(&sVti[n * 16 * 64],      64, lane);
            const v16bf bvi1 = load_frag_b(&sVti[n * 16 * 64 + 32], 64, lane);
            o_r[n] = wmma_bf(ap0, bvr0, o_r[n]);
            o_r[n] = wmma_bf(ap1, bvr1, o_r[n]);
            o_i[n] = wmma_bf(ap0, bvi0, o_i[n]);
            o_i[n] = wmma_bf(ap1, bvi1, o_i[n]);
        }
    }

    // ---- normalize + store ----
#pragma unroll
    for (int j = 0; j < 8; ++j) {
        const float inv = 1.0f / l[j];
        const int row = q0 + j + ((lane & 16) ? 8 : 0);
        const size_t base = ((size_t)(b * Ll + row)) * Dd + head_off + (lane & 15);
#pragma unroll
        for (int n = 0; n < 4; ++n) {
            Or[base + n * 16] = (bf16)(o_r[n][j] * inv);
            Oi[base + n * 16] = (bf16)(o_i[n][j] * inv);
        }
    }
}

// ---------------------------------------------------------------------------
// Host-side launch
// ---------------------------------------------------------------------------
extern "C" void kernel_launch(void* const* d_in, const int* in_sizes, int n_in,
                              void* d_out, int out_size, void* d_ws, size_t ws_size,
                              hipStream_t stream) {
    (void)in_sizes; (void)n_in; (void)out_size; (void)ws_size;

    const float* zr   = (const float*)d_in[0];
    const float* zi   = (const float*)d_in[1];
    const float* wq_r = (const float*)d_in[2];
    const float* bq_r = (const float*)d_in[3];
    const float* wq_i = (const float*)d_in[4];
    const float* bq_i = (const float*)d_in[5];
    const float* wk_r = (const float*)d_in[6];
    const float* bk_r = (const float*)d_in[7];
    const float* wk_i = (const float*)d_in[8];
    const float* bk_i = (const float*)d_in[9];
    const float* wv_r = (const float*)d_in[10];
    const float* bv_r = (const float*)d_in[11];
    const float* wv_i = (const float*)d_in[12];
    const float* bv_i = (const float*)d_in[13];
    const float* wo_r = (const float*)d_in[14];
    const float* bo_r = (const float*)d_in[15];
    const float* wo_i = (const float*)d_in[16];
    const float* bo_i = (const float*)d_in[17];

    bf16* p = (bf16*)d_ws;
    bf16* zr_bf  = p; p += MD;
    bf16* zi_bf  = p; p += MD;
    bf16* wqr_bf = p; p += DD;
    bf16* wqi_bf = p; p += DD;
    bf16* wkr_bf = p; p += DD;
    bf16* wki_bf = p; p += DD;
    bf16* wvr_bf = p; p += DD;
    bf16* wvi_bf = p; p += DD;
    bf16* wor_bf = p; p += DD;
    bf16* woi_bf = p; p += DD;
    bf16* qr_bf  = p; p += MD;
    bf16* qi_bf  = p; p += MD;
    bf16* kr_bf  = p; p += MD;
    bf16* ki_bf  = p; p += MD;
    bf16* vr_bf  = p; p += MD;
    bf16* vi_bf  = p; p += MD;
    bf16* or_bf  = p; p += MD;
    bf16* oi_bf  = p; p += MD;

    auto cvt = [&](const float* s, bf16* d, size_t n) {
        const int blocks = (int)((n / 4 + 255) / 256);
        cvt_f32_to_bf16<<<blocks, 256, 0, stream>>>(s, d, (int)n);
    };
    cvt(zr, zr_bf, MD);
    cvt(zi, zi_bf, MD);
    cvt(wq_r, wqr_bf, DD);
    cvt(wq_i, wqi_bf, DD);
    cvt(wk_r, wkr_bf, DD);
    cvt(wk_i, wki_bf, DD);
    cvt(wv_r, wvr_bf, DD);
    cvt(wv_i, wvi_bf, DD);
    cvt(wo_r, wor_bf, DD);
    cvt(wo_i, woi_bf, DD);

    const dim3 gg(Mm / 128, Dd / 64);   // (32, 16)
    clin_gemm<false><<<gg, 256, 0, stream>>>(zr_bf, zi_bf, wqr_bf, wqi_bf, bq_r, bq_i, qr_bf, qi_bf);
    clin_gemm<false><<<gg, 256, 0, stream>>>(zr_bf, zi_bf, wkr_bf, wki_bf, bk_r, bk_i, kr_bf, ki_bf);
    clin_gemm<false><<<gg, 256, 0, stream>>>(zr_bf, zi_bf, wvr_bf, wvi_bf, bv_r, bv_i, vr_bf, vi_bf);

    const dim3 ga(Ll / 128, Bb * NHh);  // (16, 32)
    attn_kernel<<<ga, 256, 0, stream>>>(qr_bf, qi_bf, kr_bf, ki_bf, vr_bf, vi_bf, or_bf, oi_bf);

    float* out = (float*)d_out;
    clin_gemm<true><<<gg, 256, 0, stream>>>(or_bf, oi_bf, wor_bf, woi_bf, bo_r, bo_i,
                                            (void*)out, (void*)(out + MD));
}
